// TiteSelfAttention_12721693131196
// MI455X (gfx1250) — compile-verified, matrix operand order checked by reading, not run on hardware
//
#include <hip/hip_runtime.h>
#include <hip/hip_bf16.h>

// ---------- CDNA5 vector types for WMMA ----------
typedef __attribute__((ext_vector_type(16))) __bf16 v16bf;
typedef __attribute__((ext_vector_type(8)))  __bf16 v8bf;
typedef __attribute__((ext_vector_type(8)))  float  v8f;

#define NB   32
#define NS   512
#define ND   768
#define NH   12
#define NHD  64
#define NOL  256          // pooled query length
#define NEGV (-10000.0f)

static __device__ __forceinline__ unsigned short f2bf(float f) {
  unsigned u = __float_as_uint(f);
  u += 0x7fffu + ((u >> 16) & 1u);      // round-to-nearest-even
  return (unsigned short)(u >> 16);
}
static __device__ __forceinline__ float bf2f(unsigned short h) {
  return __uint_as_float(((unsigned)h) << 16);
}

// Load a 16x32 bf16 A/B WMMA fragment from a row-major (K-contiguous) source.
// Per ISA 7.12.2: lanes 0-15 hold K={0..7,16..23}, lanes 16-31 hold K={8..15,24..31}.
static __device__ __forceinline__ v16bf ld_frag(const unsigned short* p) {
  v8bf lo = *(const v8bf*)p;
  v8bf hi = *(const v8bf*)(p + 16);
  v16bf f;
#pragma unroll
  for (int i = 0; i < 8; ++i) { f[i] = lo[i]; f[i + 8] = hi[i]; }
  return f;
}

// =====================================================================
// Kernel 0: fp32 -> bf16 pre-convert (bandwidth bound, one pass)
// =====================================================================
__global__ __launch_bounds__(256)
void cvt_bf16(const float* __restrict__ src, unsigned short* __restrict__ dst, int n4)
{
  const int i = blockIdx.x * 256 + threadIdx.x;
  if (i >= n4) return;
  const float4 v = ((const float4*)src)[i];
  uint2 p;
  p.x = (unsigned)f2bf(v.x) | ((unsigned)f2bf(v.y) << 16);
  p.y = (unsigned)f2bf(v.z) | ((unsigned)f2bf(v.w) << 16);
  ((uint2*)dst)[i] = p;
}

// =====================================================================
// Kernel 1: QKV projection  C(16384,2304) = Xb(16384,768) * Wb^T + bias
// 128x128x32 tile, bf16 WMMA, fp32 accumulate. Tiles are staged into a
// double-buffered LDS via GLOBAL_LOAD_ASYNC_TO_LDS_B128 (ASYNCcnt),
// overlapping the next tile's fetch with the current tile's 8 WMMAs.
// Epilogue scatters bf16 Qraw(b,h,s,d)[masked], Kb(b,h,s,d), Vt(b,h,d,s).
// =====================================================================
__global__ __launch_bounds__(256)
void qkv_gemm(const unsigned short* __restrict__ Xb, const unsigned short* __restrict__ Wb,
              const float* __restrict__ bias, const unsigned char* __restrict__ am,
              unsigned short* __restrict__ Qraw, unsigned short* __restrict__ Kb,
              unsigned short* __restrict__ Vt)
{
  __shared__ unsigned short sA[2][128 * 32];   // 2 x 8KB
  __shared__ unsigned short sB[2][128 * 32];   // 2 x 8KB

  const int m0 = blockIdx.y * 128;             // row block in (B*S)
  const int n0 = blockIdx.x * 128;             // col block in 2304
  const int tid = threadIdx.x;
  const int lane = tid & 31, wid = tid >> 5;
  const int wm = wid >> 2, wn = wid & 3;       // 2 x 4 wave grid
  const int half8 = (lane < 16) ? 0 : 8;
  const int l15 = lane & 15;

  // LDS byte offsets (flat->LDS mapping: addr[31:0] is the LDS offset)
  const unsigned ldsA = (unsigned)(size_t)(const void*)&sA[0][0];
  const unsigned ldsB = (unsigned)(size_t)(const void*)&sB[0][0];

  // Each thread owns 2 16B chunks of the 128x32-bf16 A tile and 2 of B.
  // chunk = tid*2+j : row = chunk>>2, col8 = (chunk&3)*8
  auto issueTile = [&](int it, int buf) {
#pragma unroll
    for (int j = 0; j < 2; ++j) {
      const int chunk = tid * 2 + j;
      const int row = chunk >> 2;
      const int c8 = (chunk & 3) * 8;
      const unsigned lofs = (unsigned)((row * 32 + c8) * 2) + (unsigned)buf * 8192u;
      const unsigned long long ga =
          (unsigned long long)(Xb + (size_t)(m0 + row) * ND + it * 32 + c8);
      unsigned la = ldsA + lofs;
      asm volatile("global_load_async_to_lds_b128 %0, %1, off"
                   :: "v"(la), "v"(ga) : "memory");
      const unsigned long long gb =
          (unsigned long long)(Wb + (size_t)(n0 + row) * ND + it * 32 + c8);
      unsigned lb = ldsB + lofs;
      asm volatile("global_load_async_to_lds_b128 %0, %1, off"
                   :: "v"(lb), "v"(gb) : "memory");
    }
  };

  v8f acc[4][2];
#pragma unroll
  for (int mt = 0; mt < 4; ++mt)
#pragma unroll
    for (int nt = 0; nt < 2; ++nt) acc[mt][nt] = (v8f){};

  issueTile(0, 0);
  for (int it = 0; it < ND / 32; ++it) {
    const int cur = it & 1;
    if (it + 1 < ND / 32) {
      issueTile(it + 1, cur ^ 1);
      // async loads complete in order: <=4 outstanding => current tile landed
      asm volatile("s_wait_asynccnt 0x4" ::: "memory");
    } else {
      asm volatile("s_wait_asynccnt 0x0" ::: "memory");
    }
    __syncthreads();

    v16bf aF[4], bF[2];
#pragma unroll
    for (int mt = 0; mt < 4; ++mt)
      aF[mt] = ld_frag(&sA[cur][(wm * 64 + mt * 16 + l15) * 32 + half8]);
#pragma unroll
    for (int nt = 0; nt < 2; ++nt)
      bF[nt] = ld_frag(&sB[cur][(wn * 32 + nt * 16 + l15) * 32 + half8]);

#pragma unroll
    for (int mt = 0; mt < 4; ++mt)
#pragma unroll
      for (int nt = 0; nt < 2; ++nt)
        acc[mt][nt] = __builtin_amdgcn_wmma_f32_16x16x32_bf16(
            false, aF[mt], false, bF[nt], (short)0, acc[mt][nt], false, false);
    __syncthreads();   // all reads of buf `cur` done before it is overwritten
  }

  // epilogue: bias + scatter into attention layouts (bf16)
#pragma unroll
  for (int mt = 0; mt < 4; ++mt) {
#pragma unroll
    for (int nt = 0; nt < 2; ++nt) {
      const int n = n0 + wn * 32 + nt * 16;    // 16-wide tile never crosses 64-boundary
      const int part = n / ND;
      const int hr = n - part * ND;
      const int h = hr >> 6;
      const int d = (hr & 63) + l15;
      const float bv = bias[n + l15];
#pragma unroll
      for (int e = 0; e < 8; ++e) {
        const int m = m0 + wm * 64 + mt * 16 + e + half8;
        const int bb = m >> 9, s = m & 511;
        const float v = acc[mt][nt][e] + bv;
        if (part == 0) {
          const float msk = am[bb * NS + s] ? 1.f : 0.f;
          Qraw[((size_t)(bb * NH + h) * NS + s) * NHD + d] = f2bf(v * msk);
        } else if (part == 1) {
          Kb[((size_t)(bb * NH + h) * NS + s) * NHD + d] = f2bf(v);
        } else {
          Vt[((size_t)(bb * NH + h) * NHD + d) * NS + s] = f2bf(v);
        }
      }
    }
  }
}

// =====================================================================
// Kernel 2: masked mean-pool (k=2, stride=2) + new_mask
// =====================================================================
__global__ __launch_bounds__(256)
void pool_kernel(const unsigned short* __restrict__ Qraw,
                 const unsigned char* __restrict__ am,
                 unsigned short* __restrict__ Qpool, float* __restrict__ nm_out)
{
  const int idx = blockIdx.x * 256 + threadIdx.x;   // NB*NH*NOL*NHD
  const int d = idx & 63;
  const int o = (idx >> 6) & 255;
  const int h = (idx >> 14) % NH;
  const int b = idx / (64 * 256 * NH);
  const int s0 = o * 2, s1 = o * 2 + 1;
  const float n = (float)(am[b * NS + s0] + am[b * NS + s1]);
  const size_t base = (size_t)(b * NH + h) * NS * NHD;
  const float q0 = bf2f(Qraw[base + (size_t)s0 * NHD + d]);
  const float q1 = bf2f(Qraw[base + (size_t)s1 * NHD + d]);
  Qpool[((size_t)(b * NH + h) * NOL + o) * NHD + d] = f2bf((q0 + q1) / fmaxf(n, 1.f));
  if (h == 0 && d == 0) nm_out[b * NOL + o] = (n != 0.f) ? 1.f : 0.f;
}

// =====================================================================
// Kernel 3: flash attention per (b,h): Sq=256, Skey=512, D=64.
// 8 waves x 32 query rows; 16 key-chunks of 32; ALiBi + pair mask.
// =====================================================================
__global__ __launch_bounds__(256)
void attn_kernel(const unsigned short* __restrict__ Qp,
                 const unsigned short* __restrict__ Kb,
                 const unsigned short* __restrict__ Vt,
                 const unsigned char* __restrict__ am,
                 float* __restrict__ out)
{
  __shared__ unsigned short sP[8 * 32 * 32];        // per-wave P staging (16 KB)

  const int bh = blockIdx.x;
  const int b = bh / NH, h = bh % NH;
  const int lane = threadIdx.x & 31, wid = threadIdx.x >> 5;
  const int q0 = wid * 32;
  const int half8 = (lane < 16) ? 0 : 8;
  const int l15 = lane & 15;
  const float slope = (h < 8) ? exp2f(-(float)(h + 1))
                              : exp2f(-0.5f * (float)(2 * (h - 8) + 1));
  const unsigned char* amb = am + b * NS;

  // per-row pooled-query validity (new_mask)
  float nmq[2][8];
#pragma unroll
  for (int mt = 0; mt < 2; ++mt)
#pragma unroll
    for (int r = 0; r < 8; ++r) {
      const int q = q0 + mt * 16 + r + half8;
      nmq[mt][r] = (amb[2 * q] | amb[2 * q + 1]) ? 1.f : 0.f;
    }

  // Q fragments (held for the whole kernel): [mtile][kstep]
  const unsigned short* Qbase = Qp + (size_t)bh * NOL * NHD;
  v16bf aQ[2][2];
#pragma unroll
  for (int mt = 0; mt < 2; ++mt)
#pragma unroll
    for (int ks = 0; ks < 2; ++ks)
      aQ[mt][ks] = ld_frag(Qbase + (size_t)(q0 + mt * 16 + l15) * NHD + ks * 32 + half8);

  v8f O[2][4];
#pragma unroll
  for (int mt = 0; mt < 2; ++mt)
#pragma unroll
    for (int nd = 0; nd < 4; ++nd) O[mt][nd] = (v8f){};
  float rm[2][8], rs[2][8];
#pragma unroll
  for (int mt = 0; mt < 2; ++mt)
#pragma unroll
    for (int r = 0; r < 8; ++r) { rm[mt][r] = -1e30f; rs[mt][r] = 0.f; }

  const unsigned short* Kbase = Kb + (size_t)bh * NS * NHD;
  const unsigned short* Vbase = Vt + (size_t)bh * NHD * NS;
  unsigned short* myP = &sP[wid * 32 * 32];

  for (int c = 0; c < NS / 32; ++c) {
    const int kbase = c * 32;

    // ---- scores S = Q K^T -------------------------------------------
    v16bf bK[2][2];
#pragma unroll
    for (int nt = 0; nt < 2; ++nt)
#pragma unroll
      for (int ks = 0; ks < 2; ++ks)
        bK[nt][ks] = ld_frag(Kbase + (size_t)(kbase + nt * 16 + l15) * NHD + ks * 32 + half8);

    v8f S[2][2];
#pragma unroll
    for (int mt = 0; mt < 2; ++mt)
#pragma unroll
      for (int nt = 0; nt < 2; ++nt) {
        v8f s = (v8f){};
        s = __builtin_amdgcn_wmma_f32_16x16x32_bf16(false, aQ[mt][0], false, bK[nt][0], (short)0, s, false, false);
        s = __builtin_amdgcn_wmma_f32_16x16x32_bf16(false, aQ[mt][1], false, bK[nt][1], (short)0, s, false, false);
        S[mt][nt] = s;
      }

    float amk[2];
#pragma unroll
    for (int nt = 0; nt < 2; ++nt)
      amk[nt] = amb[kbase + nt * 16 + l15] ? 1.f : 0.f;

    // ---- online softmax update --------------------------------------
#pragma unroll
    for (int mt = 0; mt < 2; ++mt) {
#pragma unroll
      for (int r = 0; r < 8; ++r) {
        const int q = q0 + mt * 16 + r + half8;
        float sv[2], tmax = -1e30f;
#pragma unroll
        for (int nt = 0; nt < 2; ++nt) {
          const int key = kbase + nt * 16 + l15;
          const float biasv = (nmq[mt][r] * amk[nt] != 0.f) ? 0.f : NEGV;
          const float v = S[mt][nt][r] * 0.125f + biasv - slope * fabsf((float)(key - q));
          sv[nt] = v;
          tmax = fmaxf(tmax, v);
        }
#pragma unroll
        for (int off = 1; off < 16; off <<= 1)
          tmax = fmaxf(tmax, __shfl_xor(tmax, off, 32));
        const float newm = fmaxf(rm[mt][r], tmax);
        const float corr = __expf(rm[mt][r] - newm);
        rm[mt][r] = newm;
        float psum = 0.f;
        unsigned short pb[2];
#pragma unroll
        for (int nt = 0; nt < 2; ++nt) {
          const float p = __expf(sv[nt] - newm);
          psum += p;
          pb[nt] = f2bf(p);
        }
#pragma unroll
        for (int off = 1; off < 16; off <<= 1)
          psum += __shfl_xor(psum, off, 32);
        rs[mt][r] = rs[mt][r] * corr + psum;
#pragma unroll
        for (int nd = 0; nd < 4; ++nd) O[mt][nd][r] *= corr;
        const int lrow = mt * 16 + r + half8;
        myP[lrow * 32 + l15]      = pb[0];
        myP[lrow * 32 + 16 + l15] = pb[1];
      }
    }

    // wave-local LDS RAW: make P stores visible before fragment reads
    asm volatile("s_wait_dscnt 0" ::: "memory");

    v16bf aP[2];
#pragma unroll
    for (int mt = 0; mt < 2; ++mt)
      aP[mt] = ld_frag(&myP[(mt * 16 + l15) * 32 + half8]);

    // ---- O += P V (V pre-transposed: (d, s)) ------------------------
#pragma unroll
    for (int nd = 0; nd < 4; ++nd) {
      v16bf bV = ld_frag(Vbase + (size_t)(nd * 16 + l15) * NS + kbase + half8);
#pragma unroll
      for (int mt = 0; mt < 2; ++mt)
        O[mt][nd] = __builtin_amdgcn_wmma_f32_16x16x32_bf16(
            false, aP[mt], false, bV, (short)0, O[mt][nd], false, false);
    }
  }

  // ---- normalize + store out(b, o, h*64+d) fp32 ---------------------
  float* ob = out + (size_t)b * NOL * ND + (size_t)h * NHD;
#pragma unroll
  for (int mt = 0; mt < 2; ++mt)
#pragma unroll
    for (int nd = 0; nd < 4; ++nd)
#pragma unroll
      for (int e = 0; e < 8; ++e) {
        const int q = q0 + mt * 16 + e + half8;
        ob[(size_t)q * ND + nd * 16 + l15] = O[mt][nd][e] / rs[mt][e];
      }
}

// =====================================================================
extern "C" void kernel_launch(void* const* d_in, const int* in_sizes, int n_in,
                              void* d_out, int out_size, void* d_ws, size_t ws_size,
                              hipStream_t stream) {
  const float* X            = (const float*)d_in[0];          // (32,512,768)
  const unsigned char* am   = (const unsigned char*)d_in[1];  // (32,512) bool
  const float* W            = (const float*)d_in[2];          // (2304,768)
  const float* bias         = (const float*)d_in[3];          // (2304,)

  float* out_f  = (float*)d_out;                              // out (32,256,768)
  float* nm_out = out_f + (size_t)NB * NOL * ND;              // new_mask (32,256)

  // workspace carve-up (bf16 buffers). Qpool aliases Xb (dead after GEMM).
  char* ws = (char*)d_ws;
  const size_t qkvElems = (size_t)NB * NH * NS * NHD;         // 12,582,912
  const size_t xElems   = (size_t)NB * NS * ND;               // 12,582,912
  const size_t wElems   = (size_t)3 * ND * ND;                //  1,769,472
  unsigned short* Qraw  = (unsigned short*)(ws);
  unsigned short* Kb    = (unsigned short*)(ws + qkvElems * 2);
  unsigned short* Vt    = (unsigned short*)(ws + qkvElems * 4);
  unsigned short* Xb    = (unsigned short*)(ws + qkvElems * 6);
  unsigned short* Wb    = (unsigned short*)(ws + qkvElems * 6 + xElems * 2);
  unsigned short* Qpool = Xb;                                  // alias: Xb dead post-GEMM

  cvt_bf16<<<(int)(xElems / 4 / 256), 256, 0, stream>>>(X, Xb, (int)(xElems / 4));
  cvt_bf16<<<(int)(wElems / 4 / 256), 256, 0, stream>>>(W, Wb, (int)(wElems / 4));

  qkv_gemm<<<dim3(3 * ND / 128, NB * NS / 128), 256, 0, stream>>>(
      Xb, Wb, bias, am, Qraw, Kb, Vt);

  const int poolElems = NB * NH * NOL * NHD;                  // 6,291,456
  pool_kernel<<<poolElems / 256, 256, 0, stream>>>(Qraw, am, Qpool, nm_out);

  attn_kernel<<<NB * NH, 256, 0, stream>>>(Qpool, Kb, Vt, am, out_f);
}
